// get_fc_pix_discriminator_1x1_solo_63891933495359
// MI455X (gfx1250) — matrix operand order repaired and checked by reading, not compiled, not run on hardware
//
#include <hip/hip_runtime.h>
#include <hip/hip_bf16.h>
#include <math.h>

typedef _Float16 half_t;
typedef __attribute__((ext_vector_type(8)))  _Float16 v8h;
typedef __attribute__((ext_vector_type(16))) _Float16 v16h;
typedef __attribute__((ext_vector_type(8)))  float    v8f;

#define LEAKK 0.2f

constexpr int BATCH = 4;
constexpr int CIN   = 19;
constexpr int HH    = 256;
constexpr int WWID  = 512;
constexpr int HW    = HH * WWID;        // 131072 (power of two)
constexpr int NPIX  = BATCH * HW;       // 524288

constexpr int D1 = 64, D2 = 128, D3 = 256, D4 = 512;
constexpr int K1 = 32;                  // input channels padded 19 -> 32

// Padded LDS activation strides: +8 halves (16B) makes the per-lane bank
// advance == 4 (mod 16) dwords for every layer width -> conflict-free ds_load_b128.
constexpr int S0 = K1 + 8;              // 40
constexpr int S1 = D1 + 8;              // 72
constexpr int S2 = D2 + 8;              // 136
constexpr int S3 = D3 + 8;              // 264

// Fragment-major (swizzled) f16 weight table offsets, in half elements.
// Each B fragment (16x32, one WMMA K-chunk x N-tile) = 512 halves, stored as
// 16 contiguous halves per lane -> fully coalesced 1KB wave loads.
constexpr int O_W1  = 0;                // 64x32    = 2048
constexpr int O_W2  = O_W1 + D1 * K1;   // 2048   + 128x64  = 8192
constexpr int O_W3  = O_W2 + D2 * D1;   // 10240  + 256x128 = 32768
constexpr int O_W4  = O_W3 + D3 * D2;   // 43008  + 512x256 = 131072
constexpr int W_TOT = O_W4 + D4 * D3;   // 174080 halves = 340 KB

constexpr int MT    = 4;                // M-tiles (16 px each) per wave
constexpr int PIXB  = MT * 16;          // 64 pixels per (single-wave) block
constexpr int BLOCK = 32;

// ---------------------------------------------------------------------------
// Prep: f32 weights -> f16, swizzled to per-fragment/per-lane order.
// Fragment f = nt*(KD/32)+kt ; within: lane l holds B[N=l%16][K=(j/8)*16+(l/16)*8+j%8].
// Also zeroes the scalar output (runs every launch => graph-replay safe).
// ---------------------------------------------------------------------------
__device__ __forceinline__ void swz_store(const float* __restrict__ W,
                                          half_t* __restrict__ dst,
                                          int i, int KD, int realK) {
    const int f   = i >> 9;             // 512 halves per fragment
    const int rem = i & 511;
    const int l   = rem >> 4;
    const int j   = rem & 15;
    const int KT  = KD >> 5;
    const int nt  = f / KT;
    const int kt  = f - nt * KT;
    const int n   = nt * 16 + (l & 15);
    const int k   = kt * 32 + ((j >> 3) << 4) + ((l >> 4) << 3) + (j & 7);
    dst[i] = (k < realK) ? (half_t)W[n * realK + k] : (half_t)0.0f;
}

__global__ void prep_weights_kernel(const float* __restrict__ W1,
                                    const float* __restrict__ W2,
                                    const float* __restrict__ W3,
                                    const float* __restrict__ W4,
                                    half_t* __restrict__ ws,
                                    float* __restrict__ out) {
    int i = blockIdx.x * blockDim.x + threadIdx.x;
    if (i == 0) out[0] = 0.0f;
    if (i >= W_TOT) return;
    if (i < O_W2)      swz_store(W1, ws + O_W1, i - O_W1, K1, CIN);
    else if (i < O_W3) swz_store(W2, ws + O_W2, i - O_W2, D1, D1);
    else if (i < O_W4) swz_store(W3, ws + O_W3, i - O_W3, D2, D2);
    else               swz_store(W4, ws + O_W4, i - O_W4, D3, D3);
}

// ---------------------------------------------------------------------------
// One (64 px x KD -> ND) layer: B fragment loaded once per (nt,kt) and reused
// across the 4 M-tiles (4 WMMAs per weight fetch).
// ---------------------------------------------------------------------------
template<int KD, int ND, int SIN, int SOUT>
__device__ __forceinline__ void layer64(const half_t* __restrict__ inB,
                                        const half_t* __restrict__ Wg,
                                        const float*  __restrict__ bias,
                                        half_t* __restrict__ outB, int lane) {
    const int row = lane & 15;
    const int sel = lane >> 4;
    constexpr int KT = KD / 32;
    for (int nt = 0; nt < ND / 16; ++nt) {
        const int nb = nt * 16;
        v8f acc[MT];
#pragma unroll
        for (int mt = 0; mt < MT; ++mt) acc[mt] = (v8f){};
#pragma unroll
        for (int kt = 0; kt < KT; ++kt) {
            const int k0 = kt * 32;
            // B fragment: coalesced 32B/lane from the swizzled table
            const half_t* wp = Wg + (((nt * KT + kt) << 9) + (lane << 4));
            v8h blo = *(const v8h*)(wp);
            v8h bhi = *(const v8h*)(wp + 8);
            v16h b = __builtin_shufflevector(blo, bhi,
                         0,1,2,3,4,5,6,7,8,9,10,11,12,13,14,15);
#pragma unroll
            for (int mt = 0; mt < MT; ++mt) {
                const half_t* ap = inB + (mt * 16 + row) * SIN + k0 + sel * 8;
                v8h alo = *(const v8h*)(ap);
                v8h ahi = *(const v8h*)(ap + 16);
                v16h a = __builtin_shufflevector(alo, ahi,
                             0,1,2,3,4,5,6,7,8,9,10,11,12,13,14,15);
                acc[mt] = __builtin_amdgcn_wmma_f32_16x16x32_f16(
                              false, a, false, b, (short)0, acc[mt], false, false);
            }
        }
        const float bv = bias[nb + row];
#pragma unroll
        for (int mt = 0; mt < MT; ++mt)
#pragma unroll
            for (int r = 0; r < 8; ++r) {   // D layout: M = r + 8*sel, N = row
                float h = acc[mt][r] + bv;
                h = (h >= 0.0f) ? h : (LEAKK * h);
                outB[(mt * 16 + r + 8 * sel) * SOUT + nb + row] = (half_t)h;
            }
    }
    asm volatile("s_wait_dscnt 0" ::: "memory");   // same-wave LDS RAW fence
}

// ---------------------------------------------------------------------------
// Main: one wave per block pushes its private 64-pixel stripe through the MLP.
// ---------------------------------------------------------------------------
__global__ __launch_bounds__(BLOCK)
void pixmlp_bce_kernel(const float* __restrict__ x, const float* __restrict__ lbl,
                       const float* __restrict__ b1, const float* __restrict__ b2,
                       const float* __restrict__ b3, const float* __restrict__ b4,
                       const float* __restrict__ W5, const float* __restrict__ b5,
                       const half_t* __restrict__ wts, float* __restrict__ out) {
    __shared__ half_t bufA[PIXB * S2];   // 64x136 halves: in(64x40) / mid(64x136)
    __shared__ half_t bufB[PIXB * S3];   // 64x264 halves: mid(64x72) / mid(64x264)

    const int lane    = threadIdx.x & 31;
    const int pixBase = blockIdx.x * PIXB;

    // Stage 64 pixels x 19 channels (pad to 32) into LDS as f16; consecutive
    // lanes hit consecutive pixels -> coalesced global_load_b32.
    for (int s = lane; s < PIXB * K1; s += 32) {
        const int c = s >> 6, m = s & 63;
        float v = 0.0f;
        if (c < CIN) {
            const int p  = pixBase + m;
            const int b  = p >> 17;          // / HW
            const int hw = p & (HW - 1);     // % HW
            v = x[(b * CIN + c) * HW + hw];
        }
        bufA[m * S0 + c] = (half_t)v;
    }
    asm volatile("s_wait_dscnt 0" ::: "memory");

    __builtin_prefetch(wts + O_W1, 0, 1);
    layer64<K1, D1, S0, S1>(bufA, wts + O_W1, b1, bufB, lane);   //  19 ->  64
    __builtin_prefetch(wts + O_W3, 0, 1);
    layer64<D1, D2, S1, S2>(bufB, wts + O_W2, b2, bufA, lane);   //  64 -> 128
    __builtin_prefetch(wts + O_W4, 0, 1);
    layer64<D2, D3, S2, S3>(bufA, wts + O_W3, b3, bufB, lane);   // 128 -> 256

    // Layer 4 (256->512) with layer 5 (512->1) fused into the epilogue.
    const int row = lane & 15, sel = lane >> 4;
    float zacc[MT][8];
#pragma unroll
    for (int mt = 0; mt < MT; ++mt)
#pragma unroll
        for (int r = 0; r < 8; ++r) zacc[mt][r] = 0.0f;

    const half_t* W4g = wts + O_W4;
    constexpr int KT4 = D3 / 32;
    for (int nt = 0; nt < D4 / 16; ++nt) {
        const int nb = nt * 16;
        v8f acc[MT];
#pragma unroll
        for (int mt = 0; mt < MT; ++mt) acc[mt] = (v8f){};
#pragma unroll
        for (int kt = 0; kt < KT4; ++kt) {
            const int k0 = kt * 32;
            const half_t* wp = W4g + (((nt * KT4 + kt) << 9) + (lane << 4));
            v8h blo = *(const v8h*)(wp);
            v8h bhi = *(const v8h*)(wp + 8);
            v16h b = __builtin_shufflevector(blo, bhi,
                         0,1,2,3,4,5,6,7,8,9,10,11,12,13,14,15);
#pragma unroll
            for (int mt = 0; mt < MT; ++mt) {
                const half_t* ap = bufB + (mt * 16 + row) * S3 + k0 + sel * 8;
                v8h alo = *(const v8h*)(ap);
                v8h ahi = *(const v8h*)(ap + 16);
                v16h a = __builtin_shufflevector(alo, ahi,
                             0,1,2,3,4,5,6,7,8,9,10,11,12,13,14,15);
                acc[mt] = __builtin_amdgcn_wmma_f32_16x16x32_f16(
                              false, a, false, b, (short)0, acc[mt], false, false);
            }
        }
        const float bv  = b4[nb + row];
        const float w5v = W5[nb + row];
#pragma unroll
        for (int mt = 0; mt < MT; ++mt)
#pragma unroll
            for (int r = 0; r < 8; ++r) {
                float h = acc[mt][r] + bv;
                h = (h >= 0.0f) ? h : (LEAKK * h);
                zacc[mt][r] = fmaf(h, w5v, zacc[mt][r]);
            }
    }

    // Sum the N dimension across each 16-lane half (xor stays in the half).
#pragma unroll
    for (int mt = 0; mt < MT; ++mt)
#pragma unroll
        for (int r = 0; r < 8; ++r) {
#pragma unroll
            for (int off = 1; off < 16; off <<= 1)
                zacc[mt][r] += __shfl_xor(zacc[mt][r], off, 32);
        }

    if (row == 0) {                          // lanes 0,16: M = mt*16 + 8*sel + r
        const float bias5 = b5[0];
        float s = 0.0f;
#pragma unroll
        for (int mt = 0; mt < MT; ++mt)
#pragma unroll
            for (int r = 0; r < 8; ++r) {
                const int   p = pixBase + mt * 16 + r + 8 * sel;
                const float z = zacc[mt][r] + bias5;
                const float t = lbl[p];
                s += fmaxf(z, 0.0f) - z * t + log1pf(expf(-fabsf(z)));
            }
        atomicAdd(out, s * (1.0f / (float)NPIX));
    }
}

// ---------------------------------------------------------------------------
// Launch
// ---------------------------------------------------------------------------
extern "C" void kernel_launch(void* const* d_in, const int* in_sizes, int n_in,
                              void* d_out, int out_size, void* d_ws, size_t ws_size,
                              hipStream_t stream) {
    (void)in_sizes; (void)n_in; (void)out_size; (void)ws_size;
    const float* x   = (const float*)d_in[0];
    const float* lbl = (const float*)d_in[1];
    const float* W1  = (const float*)d_in[2];
    const float* b1  = (const float*)d_in[3];
    const float* W2  = (const float*)d_in[4];
    const float* b2  = (const float*)d_in[5];
    const float* W3  = (const float*)d_in[6];
    const float* b3  = (const float*)d_in[7];
    const float* W4  = (const float*)d_in[8];
    const float* b4  = (const float*)d_in[9];
    const float* W5  = (const float*)d_in[10];
    const float* b5  = (const float*)d_in[11];
    float*  out = (float*)d_out;
    half_t* wts = (half_t*)d_ws;

    const int prepBlocks = (W_TOT + 255) / 256;
    prep_weights_kernel<<<prepBlocks, 256, 0, stream>>>(W1, W2, W3, W4, wts, out);

    const int grid = NPIX / PIXB;            // 8192 single-wave blocks
    pixmlp_bce_kernel<<<grid, BLOCK, 0, stream>>>(
        x, lbl, b1, b2, b3, b4, W5, b5, wts, out);
}
